// ALSHAlexNet_26645977104465
// MI455X (gfx1250) — matrix operand order repaired and checked by reading, not compiled, not run on hardware
//
#include <hip/hip_runtime.h>
#include <hip/hip_bf16.h>

// ALSH-AlexNet forward on MI455X (gfx1250, wave32).
// Convs + FCs via V_WMMA_F32_16X16X32_F16 (f16 in, f32 acc), implicit GEMM,
// fully templated geometry (constant div/mod), register-staged double-buffered
// LDS pipeline, BK=64 (8 WMMAs per K-iteration per wave).

typedef __attribute__((ext_vector_type(16))) _Float16 v16h;
typedef __attribute__((ext_vector_type(8)))  float    v8f;

union FragU {
    v16h h;
    uint4 q[2];
};

#define BK 64

// ---------------------------------------------------------------------------
// Implicit-GEMM conv, ALSH gathers folded into staging.
//   A[m][k] = W[idxOut[m]][idxIn? idxIn[ci]:ci][kh][kw]
//   B[k][col] = im2col(X), col = (n, oh, ow)
// Tile 64(M) x 128(N) x 64(K); 8 waves; wave = 16-col strip x 4 M-subtiles.
// ---------------------------------------------------------------------------
template<int NB, int CinA, int CinFull, int H, int W_, int KH, int KW,
         int STRIDE, int PAD, int CoutA, int OH, int OW, bool HAS_IDX_IN>
__global__ __launch_bounds__(256)
void conv_wmma(const float* __restrict__ X, const float* __restrict__ Wt,
               const float* __restrict__ bias,
               const long long* __restrict__ idxOut,
               const long long* __restrict__ idxIn,
               float* __restrict__ Y)
{
    constexpr int kern   = KH * KW;
    constexpr int Kdim   = CinA * kern;
    constexpr int img    = OH * OW;
    constexpr int Ncol   = NB * img;
    constexpr int KSTEPS = (Kdim + BK - 1) / BK;

    __shared__ __align__(16) _Float16 As[2][64][BK];
    __shared__ __align__(16) _Float16 Bs[2][128][BK];

    const int tid   = threadIdx.x;
    const int lane  = tid & 31;
    const int wave  = tid >> 5;
    const int half  = lane >> 4;
    const int mBase = blockIdx.y * 64;
    const int nBase = blockIdx.x * 128;
    const int row4  = tid >> 6;      // 0..3: which row group this thread stages
    const int kk    = tid & 63;      // fixed k within tile for this thread

    // Cache the idxOut gather for the 16 A-rows this thread stages.
    int coReg[16];
    #pragma unroll
    for (int j = 0; j < 16; ++j) {
        int gm = mBase + row4 + j * 4;
        coReg[j] = (gm < CoutA) ? (int)idxOut[gm] : -1;
    }

    float aReg[16];
    float bReg[32];

    // Issue global loads for K-stage k0 into registers (decode once per stage).
    auto loadGlobal = [&](int k0) {
        const int k = k0 + kk;
        const bool kv = (k < Kdim);
        int wOff = 0, ciB = 0, khB = 0, kwB = 0;
        if (kv) {
            int ci = k / kern;           // compile-time-constant divisors
            int r  = k - ci * kern;
            int kh = r / KW;
            int kw = r - kh * KW;
            ciB = ci; khB = kh; kwB = kw;
            int cf = HAS_IDX_IN ? (int)idxIn[ci] : ci;
            wOff = (cf * KH + kh) * KW + kw;
        }
        #pragma unroll
        for (int j = 0; j < 16; ++j) {
            float v = 0.0f;
            if (kv && coReg[j] >= 0)
                v = Wt[coReg[j] * (CinFull * kern) + wOff];
            aReg[j] = v;
        }
        #pragma unroll
        for (int j = 0; j < 32; ++j) {
            int col = nBase + row4 + j * 4;
            float v = 0.0f;
            if (kv && col < Ncol) {
                int nb = col / img;
                int r2 = col - nb * img;
                int oh = r2 / OW;
                int ow = r2 - oh * OW;
                int ih = oh * STRIDE - PAD + khB;
                int iw = ow * STRIDE - PAD + kwB;
                if (ih >= 0 && ih < H && iw >= 0 && iw < W_)
                    v = X[((nb * CinA + ciB) * H + ih) * W_ + iw];
            }
            bReg[j] = v;
        }
    };

    auto storeLds = [&](int buf) {
        #pragma unroll
        for (int j = 0; j < 16; ++j)
            As[buf][row4 + j * 4][kk] = (_Float16)aReg[j];
        #pragma unroll
        for (int j = 0; j < 32; ++j)
            Bs[buf][row4 + j * 4][kk] = (_Float16)bReg[j];
    };

    v8f acc[4] = {};

    auto compute = [&](int buf) {
        #pragma unroll
        for (int ks = 0; ks < 2; ++ks) {
            FragU bf;
            int nloc = (wave << 4) + (lane & 15);
            int kb   = ks * 32 + (half << 4);
            bf.q[0] = *(const uint4*)&Bs[buf][nloc][kb];
            bf.q[1] = *(const uint4*)&Bs[buf][nloc][kb + 8];
            #pragma unroll
            for (int mi = 0; mi < 4; ++mi) {
                FragU af;
                int mloc = (mi << 4) + (lane & 15);
                int ka   = ks * 32 + (half << 4);
                af.q[0] = *(const uint4*)&As[buf][mloc][ka];
                af.q[1] = *(const uint4*)&As[buf][mloc][ka + 8];
                acc[mi] = __builtin_amdgcn_wmma_f32_16x16x32_f16(
                    false, af.h, false, bf.h, (short)0, acc[mi], false, false);
            }
        }
    };

    // Software pipeline: regs one stage ahead, LDS double-buffered.
    loadGlobal(0);
    storeLds(0);
    for (int s = 0; s < KSTEPS; ++s) {
        __syncthreads();
        if (s + 1 < KSTEPS) loadGlobal((s + 1) * BK);
        compute(s & 1);
        if (s + 1 < KSTEPS) storeLds((s + 1) & 1);
    }

    // Store D + bias back to NCHW.
    int col = nBase + (wave << 4) + (lane & 15);
    if (col < Ncol) {
        int nb = col / img;
        int r2 = col - nb * img;
        #pragma unroll
        for (int mi = 0; mi < 4; ++mi) {
            int mRow = mBase + (mi << 4) + (half << 3);
            #pragma unroll
            for (int v = 0; v < 8; ++v) {
                int gm = mRow + v;
                if (gm < CoutA)
                    Y[(nb * CoutA + gm) * img + r2] = acc[mi][v] + bias[(int)idxOut[gm]];
            }
        }
    }
}

// ---------------------------------------------------------------------------
// FC GEMM: Y[64 x Nc] = A[64 x K] * Wm[K x Nc] + bias[Nc].
// Same pipeline; weight stream is prefetched 2 stages ahead (HBM-streaming).
// ---------------------------------------------------------------------------
template<int K, int Nc>
__global__ __launch_bounds__(256)
void fc_wmma(const float* __restrict__ A, const float* __restrict__ Wm,
             const float* __restrict__ bias, float* __restrict__ Y)
{
    constexpr int KSTEPS = (K + BK - 1) / BK;

    __shared__ __align__(16) _Float16 As[2][64][BK];
    __shared__ __align__(16) _Float16 Bs[2][128][BK];

    const int tid   = threadIdx.x;
    const int lane  = tid & 31;
    const int wave  = tid >> 5;
    const int half  = lane >> 4;
    const int nBase = blockIdx.x * 128;
    const int row4  = tid >> 6;
    const int kk    = tid & 63;

    float aReg[16];
    float bReg[32];

    auto loadGlobal = [&](int k0) {
        const int k = k0 + kk;
        const bool kv = (k < K);
        #pragma unroll
        for (int j = 0; j < 16; ++j) {
            int m = row4 + j * 4;
            aReg[j] = kv ? A[m * K + k] : 0.0f;
        }
        const size_t rowB = (size_t)k * Nc;
        #pragma unroll
        for (int j = 0; j < 32; ++j) {
            int c = nBase + row4 + j * 4;
            float v = 0.0f;
            if (kv && c < Nc) v = Wm[rowB + c];
            bReg[j] = v;
        }
        // Pull the weight stream 2 stages ahead into cache (global_prefetch_b8).
        if (k + BK < K) {
            int c0 = nBase + row4;
            if (c0 < Nc) __builtin_prefetch(&Wm[(size_t)(k + BK) * Nc + c0], 0, 1);
        }
    };

    auto storeLds = [&](int buf) {
        #pragma unroll
        for (int j = 0; j < 16; ++j)
            As[buf][row4 + j * 4][kk] = (_Float16)aReg[j];
        #pragma unroll
        for (int j = 0; j < 32; ++j)
            Bs[buf][row4 + j * 4][kk] = (_Float16)bReg[j];
    };

    v8f acc[4] = {};

    auto compute = [&](int buf) {
        #pragma unroll
        for (int ks = 0; ks < 2; ++ks) {
            FragU bf;
            int nloc = (wave << 4) + (lane & 15);
            int kb   = ks * 32 + (half << 4);
            bf.q[0] = *(const uint4*)&Bs[buf][nloc][kb];
            bf.q[1] = *(const uint4*)&Bs[buf][nloc][kb + 8];
            #pragma unroll
            for (int mi = 0; mi < 4; ++mi) {
                FragU af;
                int mloc = (mi << 4) + (lane & 15);
                int ka   = ks * 32 + (half << 4);
                af.q[0] = *(const uint4*)&As[buf][mloc][ka];
                af.q[1] = *(const uint4*)&As[buf][mloc][ka + 8];
                acc[mi] = __builtin_amdgcn_wmma_f32_16x16x32_f16(
                    false, af.h, false, bf.h, (short)0, acc[mi], false, false);
            }
        }
    };

    loadGlobal(0);
    storeLds(0);
    for (int s = 0; s < KSTEPS; ++s) {
        __syncthreads();
        if (s + 1 < KSTEPS) loadGlobal((s + 1) * BK);
        compute(s & 1);
        if (s + 1 < KSTEPS) storeLds((s + 1) & 1);
    }

    int c = nBase + (wave << 4) + (lane & 15);
    if (c < Nc) {
        float bv = bias[c];
        #pragma unroll
        for (int mi = 0; mi < 4; ++mi) {
            int mRow = (mi << 4) + (half << 3);
            #pragma unroll
            for (int v = 0; v < 8; ++v)
                Y[(mRow + v) * Nc + c] = acc[mi][v] + bv;
        }
    }
}

// ---------------------------------------------------------------------------
// Elementwise helpers
// ---------------------------------------------------------------------------
__global__ void maxpool3x3s2(const float* __restrict__ in, float* __restrict__ out,
                             int NC, int H, int W, int OH, int OW)
{
    int i = blockIdx.x * blockDim.x + threadIdx.x;
    int total = NC * OH * OW;
    if (i >= total) return;
    int ow = i % OW;
    int oh = (i / OW) % OH;
    int nc = i / (OW * OH);
    const float* p = in + ((size_t)nc * H + oh * 2) * W + ow * 2;
    float m = -3.402823466e+38f;
    #pragma unroll
    for (int kh = 0; kh < 3; ++kh)
        #pragma unroll
        for (int kw = 0; kw < 3; ++kw) {
            float v = p[kh * W + kw];
            m = v > m ? v : m;
        }
    out[i] = m;
}

__global__ void zero_f32(float* __restrict__ p, int n)
{
    int i = blockIdx.x * blockDim.x + threadIdx.x;
    if (i < n) p[i] = 0.0f;
}

__global__ void scatter_ch(const float* __restrict__ src, float* __restrict__ dst,
                           const long long* __restrict__ idx,
                           int Nb, int Ca, int Cf, int HW)
{
    int i = blockIdx.x * blockDim.x + threadIdx.x;
    int total = Nb * Ca * HW;
    if (i >= total) return;
    int pos = i % HW;
    int c   = (i / HW) % Ca;
    int n   = i / (HW * Ca);
    dst[((size_t)n * Cf + (int)idx[c]) * HW + pos] = src[i];
}

// ---------------------------------------------------------------------------
extern "C" void kernel_launch(void* const* d_in, const int* in_sizes, int n_in,
                              void* d_out, int out_size, void* d_ws, size_t ws_size,
                              hipStream_t stream)
{
    (void)in_sizes; (void)n_in; (void)out_size; (void)ws_size;
    const float*     x    = (const float*)d_in[0];
    const long long* idx1 = (const long long*)d_in[1];   // jnp.int64
    const long long* idx2 = (const long long*)d_in[2];
    const long long* idx3 = (const long long*)d_in[3];
    const long long* idx4 = (const long long*)d_in[4];
    const long long* idx5 = (const long long*)d_in[5];
    const float* W1 = (const float*)d_in[6];  const float* b1 = (const float*)d_in[7];
    const float* W2 = (const float*)d_in[8];  const float* b2 = (const float*)d_in[9];
    const float* W3 = (const float*)d_in[10]; const float* b3 = (const float*)d_in[11];
    const float* W4 = (const float*)d_in[12]; const float* b4 = (const float*)d_in[13];
    const float* W5 = (const float*)d_in[14]; const float* b5 = (const float*)d_in[15];
    const float* fc6w = (const float*)d_in[16]; const float* fc6b = (const float*)d_in[17];
    const float* fc7w = (const float*)d_in[18]; const float* fc7b = (const float*)d_in[19];
    const float* fc8w = (const float*)d_in[20]; const float* fc8b = (const float*)d_in[21];
    float* out = (float*)d_out;

    // Ping-pong scratch: buf0 peak = conv1 output (47.3 MiB), buf1 peak ~11.4 MiB.
    float* buf0 = (float*)d_ws;
    float* buf1 = (float*)((char*)d_ws + (size_t)52 * 1024 * 1024);
    const int Nb = 64;

    // conv1: x[64,3,227,227] -> buf0[64,64,55,55]
    { dim3 g((Nb * 55 * 55 + 127) / 128, 1);
      conv_wmma<64, 3, 3, 227, 227, 11, 11, 4, 0, 64, 55, 55, false>
          <<<g, 256, 0, stream>>>(x, W1, b1, idx1, nullptr, buf0); }
    // pool1 -> buf1[64,64,27,27]
    { int total = Nb * 64 * 27 * 27;
      maxpool3x3s2<<<(total + 255) / 256, 256, 0, stream>>>(buf0, buf1, Nb * 64, 55, 55, 27, 27); }
    // conv2 -> buf0[64,170,27,27]
    { dim3 g((Nb * 27 * 27 + 127) / 128, (170 + 63) / 64);
      conv_wmma<64, 64, 96, 27, 27, 5, 5, 1, 2, 170, 27, 27, true>
          <<<g, 256, 0, stream>>>(buf1, W2, b2, idx2, idx1, buf0); }
    // pool2 -> buf1[64,170,13,13]
    { int total = Nb * 170 * 13 * 13;
      maxpool3x3s2<<<(total + 255) / 256, 256, 0, stream>>>(buf0, buf1, Nb * 170, 27, 27, 13, 13); }
    // conv3 -> buf0[64,256,13,13]
    { dim3 g((Nb * 169 + 127) / 128, 4);
      conv_wmma<64, 170, 256, 13, 13, 3, 3, 1, 1, 256, 13, 13, true>
          <<<g, 256, 0, stream>>>(buf1, W3, b3, idx3, idx2, buf0); }
    // conv4 -> buf1[64,256,13,13]
    { dim3 g((Nb * 169 + 127) / 128, 4);
      conv_wmma<64, 256, 384, 13, 13, 3, 3, 1, 1, 256, 13, 13, true>
          <<<g, 256, 0, stream>>>(buf0, W4, b4, idx4, idx3, buf1); }
    // conv5 -> buf0[64,170,13,13]
    { dim3 g((Nb * 169 + 127) / 128, 3);
      conv_wmma<64, 256, 384, 13, 13, 3, 3, 1, 1, 170, 13, 13, true>
          <<<g, 256, 0, stream>>>(buf1, W5, b5, idx5, idx4, buf0); }
    // pool5 -> buf1[64,170,6,6]
    { int total = Nb * 170 * 36;
      maxpool3x3s2<<<(total + 255) / 256, 256, 0, stream>>>(buf0, buf1, Nb * 170, 13, 13, 6, 6); }
    // zero-fill t = buf0[64,256,6,6], scatter active channels back
    { int total = Nb * 9216;
      zero_f32<<<(total + 255) / 256, 256, 0, stream>>>(buf0, total); }
    { int total = Nb * 170 * 36;
      scatter_ch<<<(total + 255) / 256, 256, 0, stream>>>(buf1, buf0, idx5, Nb, 170, 256, 36); }
    // fc6: [64,9216] x [9216,4096] -> buf1
    { dim3 g((4096 + 127) / 128);
      fc_wmma<9216, 4096><<<g, 256, 0, stream>>>(buf0, fc6w, fc6b, buf1); }
    // fc7: -> buf0
    { dim3 g((4096 + 127) / 128);
      fc_wmma<4096, 4096><<<g, 256, 0, stream>>>(buf1, fc7w, fc7b, buf0); }
    // fc8: -> d_out [64,1000]
    { dim3 g((1000 + 127) / 128);
      fc_wmma<4096, 1000><<<g, 256, 0, stream>>>(buf0, fc8w, fc8b, out); }
}
